// VectorQuantizer_38783554683433
// MI455X (gfx1250) — matrix operand order repaired and checked by reading, not compiled, not run on hardware
//
#include <hip/hip_runtime.h>
#include <hip/hip_bf16.h>

typedef __attribute__((ext_vector_type(16))) __bf16 v16bf;
typedef __attribute__((ext_vector_type(8)))  float  v8f;

#define D_DIM 256
#define K_DIM 1024
#define ROWS_PER_WAVE 32
#define ROWS_PER_BLOCK 128
#define MAIN_THREADS 128

// ---------------------------------------------------------------------------
// Kernel 1: codebook prep — convert to bf16, compute per-row squared norms.
// grid = K_DIM blocks, 256 threads (one per D element).
// ---------------------------------------------------------------------------
__global__ __launch_bounds__(256) void vq_prep(const float* __restrict__ cb,
                                               __bf16* __restrict__ cb_bf,
                                               float* __restrict__ cbn2) {
  __shared__ float s[256];
  const int k = blockIdx.x;
  const int d = threadIdx.x;
  const float v = cb[(size_t)k * D_DIM + d];
  cb_bf[(size_t)k * D_DIM + d] = (__bf16)v;
  s[d] = v * v;
  __syncthreads();
  for (int off = 128; off > 0; off >>= 1) {
    if (d < off) s[d] += s[d + off];
    __syncthreads();
  }
  if (d == 0) cbn2[k] = s[0];
}

// ---------------------------------------------------------------------------
// Kernel 2: main — WMMA distance matmul + argmin + gather + loss partials.
// grid = N/128 blocks, 128 threads (4 waves, each wave owns 32 rows = 2 A
// tiles that share the same B fragments: 16 WMMAs per 16 b128 loads).
// ---------------------------------------------------------------------------
__global__ __launch_bounds__(MAIN_THREADS) void vq_main(
    const float*  __restrict__ z,        // N x D f32
    const float*  __restrict__ mask,     // N
    const float*  __restrict__ cb_f32,   // K x D f32
    const __bf16* __restrict__ cb_bf,    // K x D bf16 (ws)
    const float*  __restrict__ cbn2,     // K        (ws)
    float* __restrict__ out_q,           // N x D
    float* __restrict__ out_idx,         // N (as float)
    float* __restrict__ part_loss,       // nblocks
    float* __restrict__ part_nv)         // nblocks
{
  __shared__ float red_dist[4][ROWS_PER_WAVE][16];
  __shared__ int   red_idx [4][ROWS_PER_WAVE][16];
  __shared__ int   row_sel[ROWS_PER_BLOCK];
  __shared__ float blk_red[MAIN_THREADS];

  const int tid    = threadIdx.x;
  const int wave   = tid >> 5;
  const int lane   = tid & 31;
  const int lane16 = lane & 15;
  const bool hi    = lane >= 16;
  const int rowbase = blockIdx.x * ROWS_PER_BLOCK + wave * ROWS_PER_WAVE;

  // ---- Load A fragments: two 16x256 z tiles (rows rowbase..+15, +16..+31),
  // bf16, contiguous k order (k = 16h + s), same as B:
  // lane<16 : a[c][0..15]=K(32c+0..15)  (row = lane)
  // lane>=16: a[c][0..15]=K(32c+16..31) (row = lane-16)
  v16bf a[16];   // a[0..7] tile0, a[8..15] tile1
#pragma unroll
  for (int t = 0; t < 2; ++t) {
    const int myrow = rowbase + 16 * t + lane16;
    const float* zr = z + (size_t)myrow * D_DIM + (hi ? 16 : 0);
#pragma unroll
    for (int c = 0; c < 8; ++c) {
      const float* p = zr + 32 * c;
#pragma unroll
      for (int e = 0; e < 8; ++e) a[8 * t + c][e] = (__bf16)p[e];
#pragma unroll
      for (int e = 0; e < 8; ++e) a[8 * t + c][8 + e] = (__bf16)p[8 + e];
    }
  }

  // ---- Sweep all K columns in 16-wide tiles, tracking running argmin.
  float md0[8], md1[8];
  int   mi0[8], mi1[8];
#pragma unroll
  for (int v = 0; v < 8; ++v) {
    md0[v] = 3.4e38f; md1[v] = 3.4e38f; mi0[v] = 0; mi1[v] = 0;
  }

  for (int j = 0; j < K_DIM / 16; ++j) {
    const int col = j * 16 + lane16;   // this lane's column (B col = cb row)
    // B layout (32x16 bf16): lane<16 holds K d0..d0+15 of its column,
    // lane>=16 holds K d0+16..d0+31 — contiguous 32B in the bf16 codebook.
    const __bf16* cbr = cb_bf + (size_t)col * D_DIM + (hi ? 16 : 0);
    const float cn = cbn2[col];
    if (j < K_DIM / 16 - 1)   // pull next codebook tile toward the WGP
      __builtin_prefetch(cb_bf + (size_t)(col + 16) * D_DIM, 0, 3);

    v8f acc0 = {}, acc1 = {};
    v16bf bcur = *(const v16bf*)(cbr);
#pragma unroll
    for (int c = 0; c < 8; ++c) {
      v16bf bnext;
      if (c < 7) bnext = *(const v16bf*)(cbr + 32 * (c + 1));  // in flight
      acc0 = __builtin_amdgcn_wmma_f32_16x16x32_bf16(
          false, a[c], false, bcur, (short)0, acc0, false, false);
      acc1 = __builtin_amdgcn_wmma_f32_16x16x32_bf16(
          false, a[8 + c], false, bcur, (short)0, acc1, false, false);
      if (c < 7) bcur = bnext;
    }
#pragma unroll
    for (int v = 0; v < 8; ++v) {
      const float d0 = cn - 2.0f * acc0[v];  // ||z||^2 term is row-constant
      const float d1 = cn - 2.0f * acc1[v];
      if (d0 < md0[v]) { md0[v] = d0; mi0[v] = col; }
      if (d1 < md1[v]) { md1[v] = d1; mi1[v] = col; }
    }
  }

  // ---- Cross-lane argmin reduce (C layout: lane<16 -> M=v, lane>=16 -> M=8+v)
#pragma unroll
  for (int v = 0; v < 8; ++v) {
    const int r = (hi ? 8 : 0) + v;
    red_dist[wave][r][lane16]      = md0[v];
    red_idx [wave][r][lane16]      = mi0[v];
    red_dist[wave][16 + r][lane16] = md1[v];
    red_idx [wave][16 + r][lane16] = mi1[v];
  }
  __syncthreads();

  {  // every lane reduces one of the wave's 32 rows
    const int r = lane;
    float bd = red_dist[wave][r][0];
    int   bi = red_idx [wave][r][0];
    for (int c = 1; c < 16; ++c) {
      const float d = red_dist[wave][r][c];
      const int  i2 = red_idx [wave][r][c];
      if (d < bd || (d == bd && i2 < bi)) { bd = d; bi = i2; }
    }
    row_sel[wave * ROWS_PER_WAVE + r] = bi;
  }
  __syncthreads();

  // ---- Gather winners, write quantized output, exact f32 loss partials.
  float lsum = 0.0f;
  const size_t gbase = (size_t)blockIdx.x * ROWS_PER_BLOCK;
  for (int e = tid; e < ROWS_PER_BLOCK * D_DIM; e += MAIN_THREADS) {
    const int row = e >> 8;          // D_DIM == 256
    const int d   = e & 255;
    const size_t grow = gbase + row;
    const float q  = cb_f32[(size_t)row_sel[row] * D_DIM + d];
    const float zv = z[grow * D_DIM + d];
    const float m  = mask[grow];
    out_q[grow * D_DIM + d] = q * m;
    const float diff = zv - q;
    lsum += m * diff * diff;
  }

  blk_red[tid] = lsum;
  __syncthreads();
  for (int off = MAIN_THREADS / 2; off > 0; off >>= 1) {
    if (tid < off) blk_red[tid] += blk_red[tid + off];
    __syncthreads();
  }
  if (tid == 0) part_loss[blockIdx.x] = blk_red[0];
  __syncthreads();

  // ---- n_valid partial + index output (one row per thread).
  const size_t g = gbase + tid;
  const float mv = mask[g];
  out_idx[g] = (mv > 0.0f) ? (float)row_sel[tid] : 0.0f;
  blk_red[tid] = mv;
  __syncthreads();
  for (int off = MAIN_THREADS / 2; off > 0; off >>= 1) {
    if (tid < off) blk_red[tid] += blk_red[tid + off];
    __syncthreads();
  }
  if (tid == 0) part_nv[blockIdx.x] = blk_red[0];
}

// ---------------------------------------------------------------------------
// Kernel 3: finalize scalar loss (deterministic serial sum of partials).
// ---------------------------------------------------------------------------
__global__ void vq_finalize(const float* __restrict__ part_loss,
                            const float* __restrict__ part_nv,
                            int nblocks, float* __restrict__ out_loss) {
  if (blockIdx.x == 0 && threadIdx.x == 0) {
    float l = 0.0f, nv = 0.0f;
    for (int i = 0; i < nblocks; ++i) { l += part_loss[i]; nv += part_nv[i]; }
    *out_loss = 0.25f * l / (nv * (float)D_DIM);
  }
}

extern "C" void kernel_launch(void* const* d_in, const int* in_sizes, int n_in,
                              void* d_out, int out_size, void* d_ws, size_t ws_size,
                              hipStream_t stream) {
  const float* z    = (const float*)d_in[0];  // (B,S,D) f32
  const float* mask = (const float*)d_in[1];  // (B,S)   f32
  const float* cb   = (const float*)d_in[2];  // (K,D)   f32

  const int N = in_sizes[1];                  // B*S tokens (32768)
  const int nblocks = N / ROWS_PER_BLOCK;     // 256

  // workspace layout
  char* ws = (char*)d_ws;
  __bf16* cb_bf   = (__bf16*)ws;                                   // K*D*2 B
  float*  cbn2    = (float*)(ws + (size_t)K_DIM * D_DIM * 2);      // K floats
  float*  ploss   = cbn2 + K_DIM;                                  // nblocks
  float*  pnv     = ploss + nblocks;                               // nblocks

  // output layout: quantized (N*D) | loss (1) | indices (N), all f32
  float* out_q    = (float*)d_out;
  float* out_loss = out_q + (size_t)N * D_DIM;
  float* out_idx  = out_loss + 1;

  vq_prep<<<K_DIM, 256, 0, stream>>>(cb, cb_bf, cbn2);
  vq_main<<<nblocks, MAIN_THREADS, 0, stream>>>(z, mask, cb, cb_bf, cbn2,
                                                out_q, out_idx, ploss, pnv);
  vq_finalize<<<1, 1, 0, stream>>>(ploss, pnv, nblocks, out_loss);
}
